// Head_37005438223129
// MI455X (gfx1250) — matrix operand order repaired and checked by reading, not compiled, not run on hardware
//
#include <hip/hip_runtime.h>
#include <hip/hip_bf16.h>

typedef __attribute__((ext_vector_type(16))) _Float16 v16h;
typedef __attribute__((ext_vector_type(8)))  _Float16 v8h;
typedef __attribute__((ext_vector_type(8)))  float    v8f;
typedef unsigned int u32x4 __attribute__((ext_vector_type(4)));
typedef int          i32x8 __attribute__((ext_vector_type(8)));
typedef int          i32x4 __attribute__((ext_vector_type(4)));

// Problem dims (fixed by the reference harness)
#define BB 4
#define TT 4096
#define EE 512
#define HH 64

// ---------------------------------------------------------------------------
// Kernel 1: repack Wq/Wk/Wv (f32 [512][64]) into f16 WMMA B-fragment order:
//   wpack[j][kk][nt][lane][h], lane = 16*g + (n%16), h -> K = (h<8 ? h : h+8) + 8*g.
// Every projection B-fragment load becomes one contiguous 32B chunk per lane.
// ---------------------------------------------------------------------------
__global__ void pack_w_kernel(const float* __restrict__ Wq,
                              const float* __restrict__ Wk,
                              const float* __restrict__ Wv,
                              _Float16* __restrict__ wpack) {
    int idx = blockIdx.x * blockDim.x + threadIdx.x;
    if (idx >= 3 * EE * HH) return;
    int j   = idx / (EE * HH);
    int rem = idx - j * (EE * HH);
    int k = rem >> 6;     // 0..511
    int n = rem & 63;     // 0..63
    const float* W = (j == 0) ? Wq : (j == 1) ? Wk : Wv;
    float v = W[k * HH + n];
    int kk = k >> 5, kr = k & 31;
    int g  = (kr >> 3) & 1, h  = (kr & 7) | ((kr >> 4) << 3);
    int nt = n >> 4,  nl = n & 15;
    int lane = g * 16 + nl;
    wpack[(((j * 16 + kk) * 4 + nt) * 32 + lane) * 16 + h] = (_Float16)v;
}

// LDS index for the TDM-loaded f32 x-tile: TDM pad_enable inserts
// pad_amount=4 dwords after every pad_interval=256 dwords, so element
// (row,col) with linear dword L = row*512+col lands at dword L + 4*(L>>8).
// -> effective row stride 520 dwords: 16B-aligned chunks + bank rotation.
__device__ __forceinline__ int xs_idx(int row, int col) {
    int L = row * 512 + col;
    return L + 4 * (L >> 8);
}

// ---------------------------------------------------------------------------
// Kernel 2: QKV projection with WMMA f32_16x16x32_f16.
//   One block = one (batch, 16-row) tile of x. The x-tile (16x512 f32) is
//   DMA'd into LDS by the Tensor Data Mover (TENSOR_LOAD_TO_LDS, TENSORcnt),
//   f32->f16 conversion happens at fragment-read time and co-executes with
//   the WMMAs. 4 waves, wave w owns output n-tile w (Q,K,V accumulators).
//   Outputs: qh [B][T][64] (pre-scaled by 1/sqrt(H)), kh [B][T][64],
//            vth [B][64][T] (V transposed for contiguous attention B-frags).
// ---------------------------------------------------------------------------
__global__ void __launch_bounds__(128)
qkv_proj_kernel(const float* __restrict__ x,
                const float* __restrict__ bq, const float* __restrict__ bk,
                const float* __restrict__ bv,
                const _Float16* __restrict__ wpack,
                _Float16* __restrict__ qh, _Float16* __restrict__ kh,
                _Float16* __restrict__ vth) {
    __shared__ __align__(128) float xs[16 * 512 + 4 * 32];  // + TDM pads
    const int tid  = threadIdx.x;
    const int wave = tid >> 5;
    const int lane = tid & 31;
    const int bm = blockIdx.x;
    const int b  = bm >> 8;            // 256 m-tiles per batch
    const int m0 = (bm & 255) << 4;

    const float* xb = x + (size_t)(b * TT + m0) * EE;

    // ---- TDM: DMA the 16x512 f32 tile into LDS (one wave issues it) ----
    if (wave == 0) {
        unsigned lds_addr = (unsigned)(uintptr_t)xs;       // low 32 bits = LDS offset
        unsigned long long ga = (unsigned long long)(uintptr_t)xb;
        u32x4 g0;
        g0[0] = 1u;                                        // count=1 (valid, user mode)
        g0[1] = lds_addr;                                  // D# lds_addr (bytes)
        g0[2] = (unsigned)(ga & 0xffffffffu);              // global_addr[31:0]
        g0[3] = (unsigned)((ga >> 32) & 0x1ffffffu)        // global_addr[56:32]
              | 0x80000000u;                               // type=2 ("image")
        i32x8 g1;
        g1[0] = (2 << 16)                                  // data_size = 4B
              | (1 << 20)                                  // pad_enable
              | (7 << 22)                                  // pad_interval = 256 dwords
              | (3 << 25);                                 // pad_amount  = 4 dwords
        g1[1] = (512 & 0xffff) << 16;                      // tensor_dim0 = 512 (lo16)
        g1[2] = (16 << 16);                                // dim0 hi=0 | tensor_dim1 = 16
        g1[3] = (512 << 16);                               // dim1 hi=0 | tile_dim0 = 512
        g1[4] = 16;                                        // tile_dim1 = 16, tile_dim2 = 0
        g1[5] = 512;                                       // tensor_dim0_stride = 512
        g1[6] = 0;
        g1[7] = 0;
        i32x4 z4 = {0, 0, 0, 0};                           // 2D: groups 2/3 unused
        i32x8 z8 = {0, 0, 0, 0, 0, 0, 0, 0};
        __builtin_amdgcn_tensor_load_to_lds(g0, g1, z4, z4, z8, 0);
        __builtin_amdgcn_s_wait_tensorcnt(0);
    }
    __syncthreads();

    const int nt = wave;               // 16-col output tile
    const int r  = lane & 15;
    const int g  = lane >> 4;

    v8f aq = {}, ak = {}, av = {};
#pragma unroll
    for (int kk = 0; kk < 16; ++kk) {
        // A fragment: row r, K = (h<8 ? h : h+8) + 8g within the 32-wide step.
        const int c0 = kk * 32 + 8 * g;
        float buf[16];
        *(float4*)&buf[0]  = *(const float4*)&xs[xs_idx(r, c0)];
        *(float4*)&buf[4]  = *(const float4*)&xs[xs_idx(r, c0 + 4)];
        *(float4*)&buf[8]  = *(const float4*)&xs[xs_idx(r, c0 + 16)];
        *(float4*)&buf[12] = *(const float4*)&xs[xs_idx(r, c0 + 20)];
        v16h a;
#pragma unroll
        for (int t = 0; t < 16; ++t) a[t] = (_Float16)buf[t];

        const _Float16* wb = wpack + (size_t)((kk * 4 + nt) * 32 + lane) * 16;
        v16h bqf = *(const v16h*)(wb);
        v16h bkf = *(const v16h*)(wb + 32768);   // j-stride = 16*4*32*16
        v16h bvf = *(const v16h*)(wb + 65536);

        aq = __builtin_amdgcn_wmma_f32_16x16x32_f16(false, a, false, bqf, (short)0, aq, false, false);
        ak = __builtin_amdgcn_wmma_f32_16x16x32_f16(false, a, false, bkf, (short)0, ak, false, false);
        av = __builtin_amdgcn_wmma_f32_16x16x32_f16(false, a, false, bvf, (short)0, av, false, false);
    }

    // C/D layout: element i -> (row = i + 8g, col = r) of the 16x16 tile.
    const int n = nt * 16 + r;
    const float bbq = bq[n], bbk = bk[n], bbv = bv[n];
#pragma unroll
    for (int i = 0; i < 8; ++i) {
        int row = m0 + i + 8 * g;
        size_t qi = (size_t)(b * TT + row) * HH + n;
        qh[qi] = (_Float16)((aq[i] + bbq) * 0.125f);   // fold scale = H^-0.5
        kh[qi] = (_Float16)(ak[i] + bbk);
        vth[(size_t)(b * HH + n) * TT + row] = (_Float16)(av[i] + bbv);
    }
}

// ---------------------------------------------------------------------------
// Kernel 3: flash attention. One wave per 16-query tile; key blocks of 32.
//   S (16x32) via 4 WMMAs, online softmax in f32 with shfl_xor row
//   reductions, P relaid C->A layout through per-wave LDS, O += P*V via
//   4 WMMAs against the pre-transposed V. Next block's K/V lines are
//   prefetched (global_prefetch_b8) while the softmax VALU work runs.
// ---------------------------------------------------------------------------
__global__ void __launch_bounds__(128)
attn_kernel(const _Float16* __restrict__ qh, const _Float16* __restrict__ kh,
            const _Float16* __restrict__ vth, float* __restrict__ out) {
    __shared__ __align__(32) _Float16 ps[4][16 * 40];
    const int tid  = threadIdx.x;
    const int wave = tid >> 5;
    const int lane = tid & 31;
    const int mt = blockIdx.x * 4 + wave;
    const int b  = mt >> 8;
    const int m0 = (mt & 255) << 4;
    const int r  = lane & 15;
    const int g  = lane >> 4;

    // Resident Q fragments: 2 k-steps across head dim 64 (rows m0..m0+15).
    v16h qa[2];
    {
        const _Float16* qrow = qh + (size_t)(b * TT + m0 + r) * HH;
#pragma unroll
        for (int s = 0; s < 2; ++s) {
            v8h lo = *(const v8h*)(qrow + s * 32 + 16 * g);
            v8h hi = *(const v8h*)(qrow + s * 32 + 16 * g + 8);
#pragma unroll
            for (int t = 0; t < 8; ++t) { qa[s][t] = lo[t]; qa[s][8 + t] = hi[t]; }
        }
    }

    v8f o0 = {}, o1 = {}, o2 = {}, o3 = {};
    float mrow[8], lrow[8];
#pragma unroll
    for (int i = 0; i < 8; ++i) { mrow[i] = -1e30f; lrow[i] = 0.f; }

    _Float16* myp = ps[wave];
    const int nblk = (m0 + 47) >> 5;   // key blocks of 32 up to the diagonal

#pragma unroll 2
    for (int jb = 0; jb < nblk; ++jb) {
        const int j0 = jb * 32;

        // ---- S = Q * K^T : two 16x16 C fragments (keys j0.. / j0+16..) ----
        v8f s0 = {}, s1 = {};
#pragma unroll
        for (int s = 0; s < 2; ++s) {
            // B = K^T fragment: col = key (lane&15), K-halves run along the
            // head dim -> contiguous 32B per lane in row-major K.
            v16h kb0 = *(const v16h*)(kh + (size_t)(b * TT + j0 + r) * HH + s * 32 + 16 * g);
            v16h kb1 = *(const v16h*)(kh + (size_t)(b * TT + j0 + 16 + r) * HH + s * 32 + 16 * g);
            s0 = __builtin_amdgcn_wmma_f32_16x16x32_f16(false, qa[s], false, kb0, (short)0, s0, false, false);
            s1 = __builtin_amdgcn_wmma_f32_16x16x32_f16(false, qa[s], false, kb1, (short)0, s1, false, false);
        }

        // ---- prefetch next key block (L2-resident lines) ----
        if (jb + 1 < nblk) {
            const int jn = j0 + 32;
            __builtin_prefetch(kh + (size_t)(b * TT + jn + r) * HH + 16 * g, 0, 1);
            __builtin_prefetch(kh + (size_t)(b * TT + jn + 16 + r) * HH + 16 * g, 0, 1);
            __builtin_prefetch(vth + (size_t)(b * HH + r) * TT + jn + 16 * g, 0, 1);
            __builtin_prefetch(vth + (size_t)(b * HH + 32 + r) * TT + jn + 16 * g, 0, 1);
        }

        // ---- causal mask (only the 1-2 diagonal blocks take this path) ----
        if (j0 + 31 > m0) {
#pragma unroll
            for (int i = 0; i < 8; ++i) {
                int q = m0 + i + 8 * g;
                int k0 = j0 + r;
                if (k0 > q)      s0[i] = -1e30f;
                if (k0 + 16 > q) s1[i] = -1e30f;
            }
        }

        // ---- online softmax (f32), row reductions across 16-lane halves ----
        float alpha[8];
#pragma unroll
        for (int i = 0; i < 8; ++i) {
            float t = fmaxf(s0[i], s1[i]);
            t = fmaxf(t, __shfl_xor(t, 1));
            t = fmaxf(t, __shfl_xor(t, 2));
            t = fmaxf(t, __shfl_xor(t, 4));
            t = fmaxf(t, __shfl_xor(t, 8));
            float mn = fmaxf(mrow[i], t);
            alpha[i] = __expf(mrow[i] - mn);
            mrow[i] = mn;
            float p0 = __expf(s0[i] - mn);
            float p1 = __expf(s1[i] - mn);
            s0[i] = p0; s1[i] = p1;
            float rs = p0 + p1;
            rs += __shfl_xor(rs, 1);
            rs += __shfl_xor(rs, 2);
            rs += __shfl_xor(rs, 4);
            rs += __shfl_xor(rs, 8);
            lrow[i] = lrow[i] * alpha[i] + rs;
        }
#pragma unroll
        for (int i = 0; i < 8; ++i) {
            o0[i] *= alpha[i]; o1[i] *= alpha[i];
            o2[i] *= alpha[i]; o3[i] *= alpha[i];
        }

        // ---- P: C-layout -> A-layout via per-wave LDS (stride 40 halves) ----
#pragma unroll
        for (int i = 0; i < 8; ++i) {
            int row = i + 8 * g;
            myp[row * 40 + r]      = (_Float16)s0[i];
            myp[row * 40 + 16 + r] = (_Float16)s1[i];
        }
        v16h pa;
        {
            v8h lo = *(const v8h*)&myp[r * 40 + 16 * g];
            v8h hi = *(const v8h*)&myp[r * 40 + 16 * g + 8];
#pragma unroll
            for (int t = 0; t < 8; ++t) { pa[t] = lo[t]; pa[8 + t] = hi[t]; }
        }

        // ---- O += P * V : V pre-transposed -> contiguous B fragments ----
        v16h vb0 = *(const v16h*)(vth + (size_t)(b * HH +  0 + r) * TT + j0 + 16 * g);
        v16h vb1 = *(const v16h*)(vth + (size_t)(b * HH + 16 + r) * TT + j0 + 16 * g);
        v16h vb2 = *(const v16h*)(vth + (size_t)(b * HH + 32 + r) * TT + j0 + 16 * g);
        v16h vb3 = *(const v16h*)(vth + (size_t)(b * HH + 48 + r) * TT + j0 + 16 * g);
        o0 = __builtin_amdgcn_wmma_f32_16x16x32_f16(false, pa, false, vb0, (short)0, o0, false, false);
        o1 = __builtin_amdgcn_wmma_f32_16x16x32_f16(false, pa, false, vb1, (short)0, o1, false, false);
        o2 = __builtin_amdgcn_wmma_f32_16x16x32_f16(false, pa, false, vb2, (short)0, o2, false, false);
        o3 = __builtin_amdgcn_wmma_f32_16x16x32_f16(false, pa, false, vb3, (short)0, o3, false, false);
    }

    // ---- epilogue: normalize and store f32 ----
#pragma unroll
    for (int i = 0; i < 8; ++i) {
        float inv = 1.0f / lrow[i];
        int row = m0 + i + 8 * g;
        float* op = out + (size_t)(b * TT + row) * HH;
        op[ 0 + r] = o0[i] * inv;
        op[16 + r] = o1[i] * inv;
        op[32 + r] = o2[i] * inv;
        op[48 + r] = o3[i] * inv;
    }
}

// ---------------------------------------------------------------------------
extern "C" void kernel_launch(void* const* d_in, const int* in_sizes, int n_in,
                              void* d_out, int out_size, void* d_ws, size_t ws_size,
                              hipStream_t stream) {
    const float* x  = (const float*)d_in[0];
    const float* Wq = (const float*)d_in[1];
    const float* bq = (const float*)d_in[2];
    const float* Wk = (const float*)d_in[3];
    const float* bk = (const float*)d_in[4];
    const float* Wv = (const float*)d_in[5];
    const float* bv = (const float*)d_in[6];
    float* out = (float*)d_out;

    // Workspace layout (all 256B-aligned offsets):
    //   wpack : 3*512*64 f16 = 196608 B
    //   qh/kh : [B][T][64] f16 = 2 MB each
    //   vth   : [B][64][T] f16 = 2 MB
    char* ws = (char*)d_ws;
    _Float16* wpack = (_Float16*)(ws);
    _Float16* qh    = (_Float16*)(ws + 196608);
    _Float16* kh    = (_Float16*)(ws + 196608 + 2097152);
    _Float16* vth   = (_Float16*)(ws + 196608 + 2 * 2097152);

    pack_w_kernel<<<(3 * EE * HH + 255) / 256, 256, 0, stream>>>(Wq, Wk, Wv, wpack);
    qkv_proj_kernel<<<BB * (TT / 16), 128, 0, stream>>>(x, bq, bk, bv, wpack, qh, kh, vth);
    attn_kernel<<<BB * (TT / 16) / 4, 128, 0, stream>>>(qh, kh, vth, out);
}
